// TopKGating_3582002725016
// MI455X (gfx1250) — matrix operand order repaired and checked by reading, not compiled, not run on hardware
//
#include <hip/hip_runtime.h>
#include <hip/hip_bf16.h>
#include <math.h>

// ---------------------------------------------------------------------------
// MoE noisy-top-k gating, fused for MI455X (gfx1250, wave32, WMMA).
// seq_len == 1 => attention weight == 1 => h = (x@Wv + bv) @ Wo + bo.
// All GEMMs via v_wmma_f32_16x16x32_bf16; one wave per workgroup owns 16 rows.
// Weights pre-packed (f32 -> bf16) into fragment-native layout in d_ws:
// each B operand is one contiguous 32B global load per lane. Two n-tiles are
// processed per A fragment (half the LDS traffic, two independent WMMA chains).
// ---------------------------------------------------------------------------

typedef __attribute__((ext_vector_type(16))) __bf16          v16bf;
typedef __attribute__((ext_vector_type(8)))  float           v8f;
typedef __attribute__((ext_vector_type(8)))  unsigned short  v8us;
typedef __attribute__((ext_vector_type(4)))  unsigned short  v4us;

#define B_ROWS   131072
#define DIM      1024
#define INNER    64     // H*HD
#define NE       64     // experts
#define TOPK     8
#define G1D      256
#define G2D      128
#define N1D      64
#define N2D      128

#define WG_ROWS  16     // one wave per workgroup
#define THREADS  32

// packed-weight element offsets inside d_ws (bf16 elements)
#define OFF_WV   0
#define OFF_WO   (OFF_WV  + DIM * INNER)
#define OFF_WG1  (OFF_WO  + INNER * DIM)
#define OFF_WN1  (OFF_WG1 + DIM * G1D)
#define OFF_WG2  (OFF_WN1 + DIM * N1D)
#define OFF_WN2  (OFF_WG2 + G1D * G2D)
#define OFF_WG3  (OFF_WN2 + INNER * N2D)
#define OFF_WN3  (OFF_WG3 + G2D * NE)

// ---- bf16 helper: native hardware convert ---------------------------------
__device__ __forceinline__ unsigned short f2us(float f) {
  __bf16 b = (__bf16)f;                 // native f32->bf16 convert (RTNE)
  unsigned short u; __builtin_memcpy(&u, &b, 2); return u;
}

// ---- WMMA fragment builders (CDNA5 16x16x32 bf16 layouts) -----------------
// A (16x32, MxK) from LDS u16 row-major [16][lda]:
//   lanes 0-15 : M=lane,    K in [k0,k0+8) and [k0+16,k0+24)
//   lanes 16-31: M=lane-16, K in [k0+8,k0+16) and [k0+24,k0+32)
// => two contiguous 16B LDS loads per lane.
__device__ __forceinline__ v16bf a_frag_lds(const unsigned short* A, int lda, int k0) {
  const int lane = threadIdx.x & 31;
  const unsigned short* p = A + (lane & 15) * lda + k0 + ((lane >> 4) << 3);
  union { v8us u[2]; v16bf b; } t;
  t.u[0] = *(const v8us*)(p);
  t.u[1] = *(const v8us*)(p + 16);
  return t.b;
}

// B (32x16, KxN) from packed global bf16: fragment (kt,nt) is 512 contiguous
// elements; lane's 16 values are one contiguous 32B global load.
__device__ __forceinline__ v16bf b_frag_packed(const unsigned short* Wp,
                                               int ntiles, int kt, int nt) {
  const int lane = threadIdx.x & 31;
  const unsigned short* p = Wp + ((size_t)(kt * ntiles + nt) << 9) + (lane << 4);
  v16bf b; __builtin_memcpy(&b, p, 32);
  return b;
}

__device__ __forceinline__ v8f wmma_bf16(v16bf a, v16bf b, v8f c) {
  return __builtin_amdgcn_wmma_f32_16x16x32_bf16(false, a, false, b,
                                                 (short)0, c, false, false);
}

// two 16x16 output tiles sharing each A fragment (independent WMMA chains)
__device__ __forceinline__ void gemm2(const unsigned short* A, int lda,
                                      const unsigned short* Wp, int ntiles,
                                      int nt, int K, v8f& c0, v8f& c1) {
  c0 = (v8f){}; c1 = (v8f){};
  const int kts = K >> 5;
#pragma unroll 4
  for (int kt = 0; kt < kts; ++kt) {
    const v16bf a = a_frag_lds(A, lda, kt << 5);
    c0 = wmma_bf16(a, b_frag_packed(Wp, ntiles, kt, nt),     c0);
    c1 = wmma_bf16(a, b_frag_packed(Wp, ntiles, kt, nt + 1), c1);
  }
}

// C/D tile store: VGPR i -> row (lane<16 ? i : 8+i), col = n0+(lane&15)
__device__ __forceinline__ void store_us(unsigned short* D, int ld, int n0, v8f c,
                                         const float* bias, bool relu) {
  const int lane = threadIdx.x & 31;
  const int col  = n0 + (lane & 15);
  const int rb   = (lane >> 4) * 8;
  const float bb = bias[col];
#pragma unroll
  for (int i = 0; i < 8; ++i) {
    float f = c[i] + bb;
    if (relu) f = fmaxf(f, 0.0f);
    D[(rb + i) * ld + col] = f2us(f);
  }
}

__device__ __forceinline__ void store_f32(float* D, int ld, int n0, v8f c,
                                          const float* bias, bool softplus_) {
  const int lane = threadIdx.x & 31;
  const int col  = n0 + (lane & 15);
  const int rb   = (lane >> 4) * 8;
  const float bb = bias[col];
#pragma unroll
  for (int i = 0; i < 8; ++i) {
    float f = c[i] + bb;
    if (softplus_) f = (f > 20.0f) ? f : log1pf(expf(f));
    D[(rb + i) * ld + col] = f;
  }
}

// full layer: D[16 x N] = act(A[16 x K] @ Wp + bias), bf16 out to LDS
__device__ __forceinline__ void layer_us(unsigned short* D, int N,
                                         const unsigned short* A, int K,
                                         const unsigned short* Wp,
                                         const float* bias, bool relu) {
  for (int nt = 0; nt < (N >> 4); nt += 2) {
    v8f c0, c1;
    gemm2(A, K, Wp, N >> 4, nt, K, c0, c1);
    store_us(D, N, (nt << 4),      c0, bias, relu);
    store_us(D, N, (nt << 4) + 16, c1, bias, relu);
  }
}

// ---------------------------------------------------------------------------
// Pack one f32 weight [K][N] into fragment-native bf16 layout:
// out[f*512 + lane*16 + i] = W[kt*32 + (i>>3)*16 + (lane>>4)*8 + (i&7)][nt*16 + (lane&15)]
// with f = kt*(N/16) + nt.
__global__ void pack_weight_kernel(const float* __restrict__ W,
                                   int K, int N, unsigned short* __restrict__ dst) {
  const int o = blockIdx.x * blockDim.x + threadIdx.x;
  if (o >= K * N) return;
  const int f    = o >> 9;
  const int r    = o & 511;
  const int lane = r >> 4;
  const int i    = r & 15;
  const int ntiles = N >> 4;
  const int kt   = f / ntiles;
  const int nt   = f - kt * ntiles;
  const int k    = (kt << 5) + ((i >> 3) << 4) + ((lane >> 4) << 3) + (i & 7);
  const int n    = (nt << 4) + (lane & 15);
  dst[o] = f2us(W[(size_t)k * N + n]);
}

// ---------------------------------------------------------------------------
__global__ __launch_bounds__(THREADS)
__attribute__((amdgpu_waves_per_eu(2)))      // cap VGPRs -> 2 waves/SIMD
void moe_gate_kernel(
    const float* __restrict__ x,
    const unsigned short* __restrict__ Wp,   // packed bf16 weights (d_ws)
    const float* __restrict__ bv,  const float* __restrict__ bo,
    const float* __restrict__ bg1, const float* __restrict__ bg2,
    const float* __restrict__ bg3, const float* __restrict__ bn1,
    const float* __restrict__ bn2, const float* __restrict__ bn3,
    float* __restrict__ out_idx, float* __restrict__ out_w)
{
  // wave-private LDS staging (bf16 as u16); 60 KB total.
  // s_h doubles as the staging buffer for the x tile (x dead once v is done).
  __shared__ unsigned short s_h [16][DIM];   // 32 KB (x tile, then h)
  __shared__ unsigned short s_v [16][INNER]; //  2 KB
  __shared__ unsigned short s_g1[16][G1D];   //  8 KB
  __shared__ unsigned short s_n1[16][N1D];   //  2 KB
  __shared__ unsigned short s_g2[16][G2D];   //  4 KB
  __shared__ unsigned short s_n2[16][N2D];   //  4 KB
  __shared__ float          s_cb[16][NE];    //  4 KB
  __shared__ float          s_ns[16][NE];    //  4 KB

  const int lane = threadIdx.x & 31;
  const int row0 = blockIdx.x * WG_ROWS;

  if (threadIdx.x == 0) {                    // warm packed weights in L2
    __builtin_prefetch(Wp + OFF_WG1, 0, 3);
    __builtin_prefetch(Wp + OFF_WO,  0, 3);
  }

  // ---- stage x tile (16 rows) -> bf16 in s_h, coalesced float4 loads ------
  unsigned short* xs = &s_h[0][0];
  for (int r = 0; r < 16; ++r) {
    const float* src = x + (size_t)(row0 + r) * DIM;
    unsigned short* drow = xs + r * DIM;
    for (int c = lane * 4; c < DIM; c += 128) {
      float4 f = *(const float4*)(src + c);
      v4us o; o.x = f2us(f.x); o.y = f2us(f.y); o.z = f2us(f.z); o.w = f2us(f.w);
      *(v4us*)(drow + c) = o;
    }
  }

  // S1: v = x @ Wv + bv                  [16 x 64],   K=1024
  layer_us(&s_v[0][0],  INNER, xs,          DIM,   Wp + OFF_WV,  bv,  false);
  // S2: h = v @ Wo + bo                  [16 x 1024], K=64   (overwrites x)
  layer_us(&s_h[0][0],  DIM,   &s_v[0][0],  INNER, Wp + OFF_WO,  bo,  false);
  // S3: g1 = relu(h @ Wg1 + bg1)         [16 x 256],  K=1024
  layer_us(&s_g1[0][0], G1D,   &s_h[0][0],  DIM,   Wp + OFF_WG1, bg1, true);
  // S4: n1 = relu(h @ Wn1 + bn1)         [16 x 64],   K=1024
  layer_us(&s_n1[0][0], N1D,   &s_h[0][0],  DIM,   Wp + OFF_WN1, bn1, true);
  // S5: g2 = relu(g1 @ Wg2 + bg2)        [16 x 128],  K=256
  layer_us(&s_g2[0][0], G2D,   &s_g1[0][0], G1D,   Wp + OFF_WG2, bg2, true);
  // S6: n2 = relu(n1 @ Wn2 + bn2)        [16 x 128],  K=64
  layer_us(&s_n2[0][0], N2D,   &s_n1[0][0], N1D,   Wp + OFF_WN2, bn2, true);
  // S7: gating = g2 @ Wg3 + bg3          [16 x 64],   K=128  (f32, no act)
  {
    v8f c0, c1;
    gemm2(&s_g2[0][0], G2D, Wp + OFF_WG3, NE >> 4, 0, G2D, c0, c1);
    store_f32(&s_cb[0][0], NE, 0,  c0, bg3, false);
    store_f32(&s_cb[0][0], NE, 16, c1, bg3, false);
    gemm2(&s_g2[0][0], G2D, Wp + OFF_WG3, NE >> 4, 2, G2D, c0, c1);
    store_f32(&s_cb[0][0], NE, 32, c0, bg3, false);
    store_f32(&s_cb[0][0], NE, 48, c1, bg3, false);
  }
  // S8: noisy = softplus(n2 @ Wn3 + bn3) [16 x 64],   K=128
  {
    v8f c0, c1;
    gemm2(&s_n2[0][0], N2D, Wp + OFF_WN3, NE >> 4, 0, N2D, c0, c1);
    store_f32(&s_ns[0][0], NE, 0,  c0, bn3, true);
    store_f32(&s_ns[0][0], NE, 16, c1, bn3, true);
    gemm2(&s_n2[0][0], N2D, Wp + OFF_WN3, NE >> 4, 2, N2D, c0, c1);
    store_f32(&s_ns[0][0], NE, 32, c0, bn3, true);
    store_f32(&s_ns[0][0], NE, 48, c1, bn3, true);
  }

  // ---- epilogue: standardize (ddof=1), combine, top-8, softmax ------------
  if (lane < 16) {
    const int m = lane;
    float* nz = &s_ns[m][0];
    float* cb = &s_cb[m][0];

    float mu = 0.0f;
    for (int e = 0; e < NE; ++e) mu += nz[e];
    mu *= (1.0f / NE);
    float var = 0.0f;
    for (int e = 0; e < NE; ++e) { float d = nz[e] - mu; var += d * d; }
    const float sd  = sqrtf(var * (1.0f / (NE - 1)));
    const float inv = 1.0f / (sd + 1e-8f);
    for (int e = 0; e < NE; ++e) cb[e] += (nz[e] - mu) * inv;

    float vals[TOPK]; int idx[TOPK];
#pragma unroll
    for (int t = 0; t < TOPK; ++t) {
      float best = -1e30f; int bi = 0;
      for (int e = 0; e < NE; ++e) {
        float v = cb[e];
        if (v > best) { best = v; bi = e; }
      }
      vals[t] = best; idx[t] = bi; cb[bi] = -1e30f;
    }
    const float mx = vals[0];
    float w[TOPK], s = 0.0f;
#pragma unroll
    for (int t = 0; t < TOPK; ++t) { w[t] = expf(vals[t] - mx); s += w[t]; }
    const float is = 1.0f / s;
    const size_t obase = (size_t)(row0 + m) * TOPK;
#pragma unroll
    for (int t = 0; t < TOPK; ++t) {
      out_idx[obase + t] = (float)idx[t];
      out_w  [obase + t] = w[t] * is;
    }
  }
}

// ---------------------------------------------------------------------------
extern "C" void kernel_launch(void* const* d_in, const int* in_sizes, int n_in,
                              void* d_out, int out_size, void* d_ws, size_t ws_size,
                              hipStream_t stream) {
  (void)in_sizes; (void)n_in; (void)out_size; (void)ws_size;
  const float* x   = (const float*)d_in[0];
  // d_in[1..4] = Wq,bq,Wk,bk -> dead under seq_len==1 (softmax of singleton == 1)
  const float* Wv  = (const float*)d_in[5];
  const float* bv  = (const float*)d_in[6];
  const float* Wo  = (const float*)d_in[7];
  const float* bo  = (const float*)d_in[8];
  const float* Wg1 = (const float*)d_in[9];
  const float* bg1 = (const float*)d_in[10];
  const float* Wg2 = (const float*)d_in[11];
  const float* bg2 = (const float*)d_in[12];
  const float* Wg3 = (const float*)d_in[13];
  const float* bg3 = (const float*)d_in[14];
  const float* Wn1 = (const float*)d_in[15];
  const float* bn1 = (const float*)d_in[16];
  const float* Wn2 = (const float*)d_in[17];
  const float* bn2 = (const float*)d_in[18];
  const float* Wn3 = (const float*)d_in[19];
  const float* bn3 = (const float*)d_in[20];

  unsigned short* Wp = (unsigned short*)d_ws;   // ~1 MB packed bf16 weights

  struct { const float* W; int K, N; int off; } packs[8] = {
    { Wv,  DIM,   INNER, OFF_WV  }, { Wo,  INNER, DIM, OFF_WO  },
    { Wg1, DIM,   G1D,   OFF_WG1 }, { Wn1, DIM,   N1D, OFF_WN1 },
    { Wg2, G1D,   G2D,   OFF_WG2 }, { Wn2, INNER, N2D, OFF_WN2 },
    { Wg3, G2D,   NE,    OFF_WG3 }, { Wn3, N2D,   NE,  OFF_WN3 },
  };
  for (int i = 0; i < 8; ++i) {
    const int total = packs[i].K * packs[i].N;
    pack_weight_kernel<<<(total + 255) / 256, 256, 0, stream>>>(
        packs[i].W, packs[i].K, packs[i].N, Wp + packs[i].off);
  }

  float* out     = (float*)d_out;
  float* out_idx = out;                               // [B, 8] indices (as float)
  float* out_w   = out + (size_t)B_ROWS * TOPK;       // [B, 8] weights

  dim3 grid(B_ROWS / WG_ROWS);
  moe_gate_kernel<<<grid, THREADS, 0, stream>>>(
      x, Wp, bv, bo, bg1, bg2, bg3, bn1, bn2, bn3, out_idx, out_w);
}
